// MyModel_87522843560112
// MI455X (gfx1250) — matrix operand order repaired and checked by reading, not compiled, not run on hardware
//
#include <hip/hip_runtime.h>

// Problem constants (from reference)
#define B_    8
#define TQ_   2048
#define TV_   2048
#define D_    128
#define KT    64                 // keys per KV tile
#define QROWS 128                // q rows per block (8 waves x 16)
#define NW    8                  // waves per block
#define NIT   (TV_ / KT)         // 32 KV iterations

// ---- dynamic LDS layout (bytes) ----
#define KS_ROWE 136              // (128+8) bf16 per K row  -> 272 B stride (conflict-free)
#define VT_ROWE 72               // (64+8)  bf16 per Vt row -> 144 B stride
#define P_ROWE  72
#define KS_ROWB (KS_ROWE * 2)
#define VT_ROWB (VT_ROWE * 2)
#define P_ROWB  (P_ROWE * 2)
#define SM_KS(buf) ((buf) * (KT * KS_ROWB))                                  // 2 x 17408
#define SM_VT(buf) (2 * KT * KS_ROWB + (buf) * (D_ * VT_ROWB))               // 2 x 18432
#define SM_P(w)    (2 * KT * KS_ROWB + 2 * D_ * VT_ROWB + (w) * (16 * P_ROWB))
#define SMEM_BYTES (2 * KT * KS_ROWB + 2 * D_ * VT_ROWB + NW * 16 * P_ROWB)  // 90112

typedef __attribute__((ext_vector_type(16))) __bf16 v16bf;
typedef __attribute__((ext_vector_type(8)))  __bf16 v8bf;
typedef __attribute__((ext_vector_type(4)))  __bf16 v4bf;
typedef __attribute__((ext_vector_type(8)))  float  v8f;
typedef __attribute__((ext_vector_type(4)))  float  v4f;
typedef __attribute__((ext_vector_type(4)))  int    v4i_t;
typedef __attribute__((ext_vector_type(8)))  int    v8i_t;

union ab_frag { v16bf v; v8bf h[2]; };

__device__ __forceinline__ __bf16 f2bf(float f) {
  union { float f; unsigned u; } a; a.f = f;
  unsigned r = a.u + 0x7FFFu + ((a.u >> 16) & 1u);   // round-to-nearest-even
  union { unsigned short s; __bf16 b; } o; o.s = (unsigned short)(r >> 16);
  return o.b;
}

// ---------------- pre-pass 1: Q,K fp32 -> bf16 (row-major) ----------------
__global__ __launch_bounds__(256, 1)
void conv_qk_kernel(const float* __restrict__ Q, const float* __restrict__ K,
                    __bf16* __restrict__ Qb, __bf16* __restrict__ Kb) {
  const long i = (long)blockIdx.x * 256 + threadIdx.x;    // float4 index
  v4f qf = *(const v4f*)(Q + i * 4);
  v4f kf = *(const v4f*)(K + i * 4);
  v4bf qb, kb;
  #pragma unroll
  for (int e = 0; e < 4; ++e) { qb[e] = f2bf(qf[e]); kb[e] = f2bf(kf[e]); }
  *(v4bf*)(Qb + i * 4) = qb;
  *(v4bf*)(Kb + i * 4) = kb;
}

// ---------------- pre-pass 2: V fp32 -> bf16 transposed [B][D][TV] ----------------
__global__ __launch_bounds__(256, 1)
void transpose_v_kernel(const float* __restrict__ V, __bf16* __restrict__ Vtb) {
  __shared__ __bf16 T[D_][KT + 8];
  const int tid = threadIdx.x;
  const int b   = blockIdx.x / (TV_ / KT);
  const int kt  = blockIdx.x % (TV_ / KT);
  const int k0  = kt * KT;
  const float* vbase = V + ((long)b * TV_ + k0) * D_;
  #pragma unroll
  for (int i = 0; i < 8; ++i) {
    int fi = tid + i * 256;
    int key = fi >> 5;
    int d4  = (fi & 31) * 4;
    v4f vf = *(const v4f*)(vbase + key * D_ + d4);
    #pragma unroll
    for (int e = 0; e < 4; ++e) T[d4 + e][key] = f2bf(vf[e]);
  }
  __syncthreads();
  const int row = tid >> 1, kh = tid & 1;                  // 2 threads per d-row
  __bf16* dst = Vtb + ((long)b * D_ + row) * TV_ + k0 + kh * 32;
  const __bf16* src = &T[row][kh * 32];
  #pragma unroll
  for (int x = 0; x < 4; ++x) *(v8bf*)(dst + x * 8) = *(const v8bf*)(src + x * 8);
}

// ---- TDM: one-instruction 2D tile DMA global->LDS with HW row padding ----
// Descriptor layout per CDNA5 ISA (2-group form, 2D).
__device__ __forceinline__ void tdm_load_2d(unsigned ldsOff, const void* gptr,
                                            unsigned tile_d0, unsigned tile_d1,
                                            unsigned stride_d0,
                                            unsigned tensor_d0, unsigned tensor_d1,
                                            unsigned pad_interval, unsigned pad_amount)
{
  unsigned long long ga = (unsigned long long)gptr;
  v4i_t g0;
  g0[0] = __builtin_amdgcn_readfirstlane(1);                       // count=1, user mode
  g0[1] = __builtin_amdgcn_readfirstlane((int)ldsOff);             // lds_addr
  g0[2] = __builtin_amdgcn_readfirstlane((int)(unsigned)ga);       // global_addr[31:0]
  g0[3] = __builtin_amdgcn_readfirstlane(
            (int)(((unsigned)(ga >> 32) & 0x01FFFFFFu) | 0x80000000u)); // addr[56:32] | type=2
  v8i_t g1;
  g1[0] = __builtin_amdgcn_readfirstlane(
            (int)((1u << 16)                    // data_size = 2 bytes (bf16)
                | (1u << 20)                    // pad_enable
                | (pad_interval << 22)          // 2^(v+1) DWORDs per row segment
                | (pad_amount << 25)));         // (v+1) DWORDs of LDS padding
  g1[1] = __builtin_amdgcn_readfirstlane((int)(tensor_d0 << 16));  // tensor_dim0[15:0]
  g1[2] = __builtin_amdgcn_readfirstlane((int)((tensor_d0 >> 16) | (tensor_d1 << 16)));
  g1[3] = __builtin_amdgcn_readfirstlane((int)((tensor_d1 >> 16) | (tile_d0 << 16)));
  g1[4] = __builtin_amdgcn_readfirstlane((int)tile_d1);            // tile_dim1 (tile_dim2=0)
  g1[5] = __builtin_amdgcn_readfirstlane((int)stride_d0);          // tensor_dim0_stride[31:0]
  g1[6] = __builtin_amdgcn_readfirstlane(0);
  g1[7] = __builtin_amdgcn_readfirstlane(0);
  asm volatile("tensor_load_to_lds %0, %1" :: "s"(g0), "s"(g1) : "memory");
}

// ---------------- main flash-attention kernel (bf16 staging via TDM) -------
__global__ __launch_bounds__(256, 1)
void fa_fwd_kernel(const __bf16* __restrict__ Qb, const __bf16* __restrict__ Kb,
                   const __bf16* __restrict__ Vtb, const float* __restrict__ scale_p,
                   const int* __restrict__ mask_q, const int* __restrict__ mask_v,
                   float* __restrict__ Out)
{
  extern __shared__ char smem[];
  const unsigned sbase = (unsigned)(size_t)smem;   // LDS aperture: low 32 bits = LDS offset

  const int tid  = threadIdx.x;
  const int w    = tid >> 5;
  const int lane = tid & 31;
  const int half = lane >> 4;
  const int lcol = lane & 15;

  const int bid = blockIdx.x;
  const int b   = bid / (TQ_ / QROWS);
  const int qt  = bid % (TQ_ / QROWS);
  const int q0  = qt * QROWS + w * 16;

  const float scal = scale_p[0];

  // One TDM pair stages a whole KV tile (issued by wave 0 only; EXEC ignored by TDM).
  auto stage = [&](int buf, int k0) {
    // K tile: 64 rows x 128 bf16, row stride 128 elems; pad 16 B per 256 B row
    tdm_load_2d(sbase + SM_KS(buf), Kb + ((long)b * TV_ + k0) * D_,
                /*tile_d0=*/D_, /*tile_d1=*/KT, /*stride_d0=*/D_,
                /*tensor_d0=*/D_, /*tensor_d1=*/(unsigned)(B_ * TV_),
                /*pad_interval=*/5 /*64 DWORDs*/, /*pad_amount=*/3 /*4 DWORDs*/);
    // Vt tile: 128 rows x 64 bf16, row stride TV elems; pad 16 B per 128 B row
    tdm_load_2d(sbase + SM_VT(buf), Vtb + (long)b * D_ * TV_ + k0,
                /*tile_d0=*/KT, /*tile_d1=*/D_, /*stride_d0=*/TV_,
                /*tensor_d0=*/TV_, /*tensor_d1=*/(unsigned)(B_ * D_),
                /*pad_interval=*/4 /*32 DWORDs*/, /*pad_amount=*/3 /*4 DWORDs*/);
  };

  // ---- load this wave's 16x128 Q stripe as 4 bf16 A-fragments ----
  ab_frag qa[4];
  {
    const __bf16* qrow = Qb + ((long)b * TQ_ + q0 + lcol) * D_;
    #pragma unroll
    for (int c = 0; c < 4; ++c) {
      const int d0 = c * 32 + half * 8;
      qa[c].h[0] = *(const v8bf*)(qrow + d0);
      qa[c].h[1] = *(const v8bf*)(qrow + d0 + 16);
    }
  }

  float m_[8], l_[8];
  v8f   o_[8] = {};
  #pragma unroll
  for (int r = 0; r < 8; ++r) { m_[r] = -INFINITY; l_[r] = 0.f; }

  if (w == 0) {
    stage(0, 0);
    __builtin_amdgcn_s_wait_tensorcnt(0);
  }
  __syncthreads();

  for (int it = 0; it < NIT; ++it) {
    const int buf = it & 1;
    const int k0  = it * KT;
    if ((it + 1 < NIT) && (w == 0)) stage(buf ^ 1, k0 + KT);  // prefetch next tile

    const __bf16* KsB = (const __bf16*)(smem + SM_KS(buf));
    const __bf16* VtB = (const __bf16*)(smem + SM_VT(buf));
    __bf16*       Pw  = (__bf16*)(smem + SM_P(w));

    // ---- S = scale * Q K^T + mask bias ----
    float s_[4][8];
    #pragma unroll
    for (int t = 0; t < 4; ++t) {
      v8f acc = {};
      #pragma unroll
      for (int c = 0; c < 4; ++c) {
        ab_frag bf;
        const __bf16* kr = KsB + (t * 16 + lcol) * KS_ROWE + c * 32 + half * 8;
        bf.h[0] = *(const v8bf*)(kr);
        bf.h[1] = *(const v8bf*)(kr + 16);
        acc = __builtin_amdgcn_wmma_f32_16x16x32_bf16(false, qa[c].v, false, bf.v,
                                                      (short)0, acc, false, false);
      }
      const int   mv   = mask_v[(long)b * TV_ + k0 + t * 16 + lcol];
      const float bias = mv ? 0.f : -1e9f;
      #pragma unroll
      for (int r = 0; r < 8; ++r) s_[t][r] = scal * acc[r] + bias;
    }

    // ---- online softmax (row reductions across 16-lane half-wave) ----
    float alpha[8];
    #pragma unroll
    for (int r = 0; r < 8; ++r) {
      float rm = fmaxf(fmaxf(s_[0][r], s_[1][r]), fmaxf(s_[2][r], s_[3][r]));
      #pragma unroll
      for (int off = 1; off < 16; off <<= 1) rm = fmaxf(rm, __shfl_xor(rm, off, 32));
      float mn = fmaxf(m_[r], rm);
      alpha[r] = __expf(m_[r] - mn);
      m_[r] = mn;
      float rs = 0.f;
      #pragma unroll
      for (int t = 0; t < 4; ++t) { float p = __expf(s_[t][r] - mn); s_[t][r] = p; rs += p; }
      #pragma unroll
      for (int off = 1; off < 16; off <<= 1) rs += __shfl_xor(rs, off, 32);
      l_[r] = l_[r] * alpha[r] + rs;
    }
    #pragma unroll
    for (int j = 0; j < 8; ++j)
      #pragma unroll
      for (int r = 0; r < 8; ++r) o_[j][r] *= alpha[r];

    // ---- spill P (bf16) to per-wave LDS: C-layout -> row-major ----
    #pragma unroll
    for (int t = 0; t < 4; ++t)
      #pragma unroll
      for (int r = 0; r < 8; ++r)
        Pw[(r + half * 8) * P_ROWE + t * 16 + lcol] = f2bf(s_[t][r]);

    // ---- O += P * V ----
    #pragma unroll
    for (int kc = 0; kc < 2; ++kc) {
      ab_frag pa;
      const __bf16* pr = Pw + lcol * P_ROWE + kc * 32 + half * 8;
      pa.h[0] = *(const v8bf*)(pr);
      pa.h[1] = *(const v8bf*)(pr + 16);
      #pragma unroll
      for (int j = 0; j < 8; ++j) {
        ab_frag vb;
        const __bf16* vr = VtB + (j * 16 + lcol) * VT_ROWE + kc * 32 + half * 8;
        vb.h[0] = *(const v8bf*)(vr);
        vb.h[1] = *(const v8bf*)(vr + 16);
        o_[j] = __builtin_amdgcn_wmma_f32_16x16x32_bf16(false, pa.v, false, vb.v,
                                                        (short)0, o_[j], false, false);
      }
    }

    if (w == 0) __builtin_amdgcn_s_wait_tensorcnt(0);  // prefetched tile has landed
    __syncthreads();                                   // everyone done with this tile
  }

  // ---- epilogue: normalize, apply query mask, store ----
  float inv[8], mq[8];
  #pragma unroll
  for (int r = 0; r < 8; ++r) {
    inv[r] = 1.0f / l_[r];
    mq[r]  = mask_q[(long)b * TQ_ + q0 + r + half * 8] ? 1.0f : 0.0f;
  }
  float* obase = Out + ((long)b * TQ_ + q0) * D_;
  #pragma unroll
  for (int j = 0; j < 8; ++j)
    #pragma unroll
    for (int r = 0; r < 8; ++r)
      obase[(r + half * 8) * D_ + j * 16 + lcol] = o_[j][r] * inv[r] * mq[r];
}

extern "C" void kernel_launch(void* const* d_in, const int* in_sizes, int n_in,
                              void* d_out, int out_size, void* d_ws, size_t ws_size,
                              hipStream_t stream) {
  (void)in_sizes; (void)n_in; (void)out_size; (void)ws_size;
  const float* Q  = (const float*)d_in[0];
  const float* V  = (const float*)d_in[1];
  const float* K  = (const float*)d_in[2];
  const float* sc = (const float*)d_in[3];
  const int*   mq = (const int*)d_in[4];
  const int*   mv = (const int*)d_in[5];
  float* out = (float*)d_out;

  // workspace partition: Qb | Kb | Vtb (bf16), 4 MiB each = 12.58 MB total
  const long nElem = (long)B_ * TQ_ * D_;
  __bf16* Qb  = (__bf16*)d_ws;
  __bf16* Kb  = Qb + nElem;
  __bf16* Vtb = Kb + nElem;

  conv_qk_kernel<<<dim3((unsigned)(nElem / 4 / 256)), dim3(256), 0, stream>>>(Q, K, Qb, Kb);
  transpose_v_kernel<<<dim3(B_ * (TV_ / KT)), dim3(256), 0, stream>>>(V, Vtb);

  fa_fwd_kernel<<<dim3(B_ * (TQ_ / QROWS)), dim3(256), SMEM_BYTES, stream>>>(
      Qb, Kb, Vtb, sc, mq, mv, out);
}
